// PointConv_74646531605090
// MI455X (gfx1250) — matrix operand order repaired and checked by reading, not compiled, not run on hardware
//
#include <hip/hip_runtime.h>
#include <hip/hip_bf16.h>
#include <math.h>

// ---- problem constants (match reference) ----
#define BATCH 4
#define NKEY  2048
#define NPT   8192
#define NBR   32
#define DIM   3
#define C_IN  16
#define W_HID 32
#define C_MID 16
#define F_HID 64
#define C_OUT 64
#define KPG   16     // keys per workgroup in fused kernel

typedef __attribute__((ext_vector_type(16))) _Float16 v16h;
typedef __attribute__((ext_vector_type(8)))  float    v8f;

// =====================================================================
// Kernel 0: convert final-MLP weights to f16 AND pre-swizzle them into
// per-lane WMMA B-fragment order:  [ntile][kblock][lane(32)][16 halves]
// half h of lane l in block (nt,kb) = W[kb*32 + (l>>4)*16 + h][nt*16 + (l&15)]
// Each lane in the fused kernel then loads one contiguous 32B v16h.
// =====================================================================
__global__ __launch_bounds__(256) void prep_weights(
    const float* __restrict__ fw1, const float* __restrict__ fw2,
    _Float16* __restrict__ gw1s, _Float16* __restrict__ gw2s) {
  int i = blockIdx.x * 256 + threadIdx.x;
  if (i < (C_IN * C_MID) * F_HID) {            // 16384 = 4 nt * 8 kb * 32 * 16
    int h    = i & 15;
    int lane = (i >> 4) & 31;
    int blk  = i >> 9;                          // 0..31
    int kb   = blk & 7;
    int nt   = blk >> 3;
    int kr   = kb * 32 + (lane >> 4) * 16 + h;
    int col  = nt * 16 + (lane & 15);
    gw1s[i] = (_Float16)fw1[kr * F_HID + col];
  }
  if (i < F_HID * C_OUT) {                      // 4096 = 4 nt * 2 kb * 32 * 16
    int h    = i & 15;
    int lane = (i >> 4) & 31;
    int blk  = i >> 9;                          // 0..7
    int kb   = blk & 1;
    int nt   = blk >> 1;
    int kr   = kb * 32 + (lane >> 4) * 16 + h;
    int col  = nt * 16 + (lane & 15);
    gw2s[i] = (_Float16)fw2[kr * C_OUT + col];
  }
}

// =====================================================================
// Kernel 1: per-key exact top-32 via radix select on float bit patterns.
// One workgroup (256 threads) per (b,k). Distances staged in LDS.
// Invalid points: dist += 1e10 (selection-equivalent to dist+max*3000,
// and selected-invalid neighbors are zero-masked downstream anyway).
// =====================================================================
__global__ __launch_bounds__(256) void knn_kernel(
    const float* __restrict__ keys, const float* __restrict__ points,
    const unsigned char* __restrict__ valid, int* __restrict__ out_idx) {
  __shared__ unsigned s_bits[NPT];      // 32 KB distance bit patterns
  __shared__ unsigned s_hist[256];
  __shared__ unsigned s_digit, s_prevcum, s_cntLess, s_cntTie;

  const int key_lin = blockIdx.x;             // b*NKEY + k
  const int b = key_lin / NKEY;
  const int tid = threadIdx.x;

  const float kx = keys[key_lin * 3 + 0];
  const float ky = keys[key_lin * 3 + 1];
  const float kz = keys[key_lin * 3 + 2];
  const float* __restrict__ pb = points + (size_t)b * NPT * 3;
  const unsigned char* __restrict__ vb = valid + (size_t)key_lin * NPT;

  for (int n = tid; n < NPT; n += 256) {
    __builtin_prefetch(pb + (n + 512) * 3, 0, 0);   // global_prefetch_b8
    float dx = pb[n * 3 + 0] - kx;
    float dy = pb[n * 3 + 1] - ky;
    float dz = pb[n * 3 + 2] - kz;
    float d = sqrtf(fmaf(dx, dx, fmaf(dy, dy, dz * dz)));
    if (!vb[n]) d += 1e10f;                         // push invalid past all valid
    s_bits[n] = __float_as_uint(d);                 // >=0 -> u32 order == float order
  }
  __syncthreads();

  // ---- 4-pass radix select for the 32nd-smallest value T ----
  unsigned pref = 0u, maskb = 0u, remaining = NBR, below = 0u;
  for (int lv = 3; lv >= 0; --lv) {
    const int sh = lv * 8;
    s_hist[tid] = 0u;
    __syncthreads();
    for (int n = tid; n < NPT; n += 256) {
      unsigned u = s_bits[n];
      if ((u & maskb) == pref) atomicAdd(&s_hist[(u >> sh) & 0xFFu], 1u);
    }
    __syncthreads();
    // inclusive Hillis-Steele scan over 256 bins
    for (int off = 1; off < 256; off <<= 1) {
      unsigned t = (tid >= off) ? s_hist[tid - off] : 0u;
      __syncthreads();
      s_hist[tid] += t;
      __syncthreads();
    }
    unsigned cum = s_hist[tid];
    unsigned cumPrev = tid ? s_hist[tid - 1] : 0u;
    if (cum >= remaining && cumPrev < remaining) { s_digit = (unsigned)tid; s_prevcum = cumPrev; }
    __syncthreads();
    pref |= (s_digit << sh);
    maskb |= (0xFFu << sh);
    remaining -= s_prevcum;   // rank within chosen bin
    below += s_prevcum;       // count strictly below prefix so far
    __syncthreads();
  }
  const unsigned T = pref;
  const unsigned tiesNeeded = remaining;  // == NBR - below

  if (tid == 0) { s_cntLess = 0u; s_cntTie = 0u; }
  __syncthreads();

  int* __restrict__ gi = out_idx + (size_t)key_lin * NBR;
  for (int n = tid; n < NPT; n += 256) {
    unsigned u = s_bits[n];
    if (u < T) {
      unsigned p = atomicAdd(&s_cntLess, 1u);
      gi[p] = n;
    } else if (u == T) {
      unsigned p = atomicAdd(&s_cntTie, 1u);
      if (p < tiesNeeded) gi[below + p] = n;
    }
  }
}

// =====================================================================
// Kernel 2: fused gather + weight-MLP + per-key einsum (WMMA) +
// final MLP as WMMA GEMMs over 16 keys. 128 threads = 4 waves.
// Fragment layouts (ISA 7.12.2, wave32):
//  A (16x32 f16): lane l -> row M=l&15; half h -> K=(l>>4)*16+h (contiguous k order chosen here)
//  B (32x16 f16): lane l -> col N=l&15; half h -> K=(l>>4)*16+h (same order as A, chosen here)
//  C/D (16x16 f32): lane l -> col N=l&15; VGPR r -> row M=r+8*(l>>4)
// =====================================================================
__global__ __launch_bounds__(128) void fused_kernel(
    const float* __restrict__ keys, const float* __restrict__ points,
    const float* __restrict__ feats, const unsigned char* __restrict__ valid,
    const float* __restrict__ ww1, const float* __restrict__ wb1,
    const float* __restrict__ ww2, const float* __restrict__ wb2,
    const _Float16* __restrict__ gw1s, const float* __restrict__ fb1,
    const _Float16* __restrict__ gw2s, const float* __restrict__ fb2,
    const int* __restrict__ idxs, float* __restrict__ out) {
  __shared__ float   s_ww1[DIM * W_HID];
  __shared__ float   s_wb1[W_HID];
  __shared__ float   s_ww2[W_HID * C_MID];
  __shared__ float   s_wb2[C_MID];
  __shared__ _Float16 s_A [KPG][C_MID][NBR];   // m^T per key (A-matrix src)
  __shared__ _Float16 s_Bt[KPG][C_IN][NBR];    // masked feats, TRANSPOSED (col-major B src)
  __shared__ _Float16 s_E [KPG][C_MID * C_IN]; // einsum result (256 per key)
  __shared__ _Float16 s_H [KPG][F_HID];        // hidden layer
  __shared__ int      s_idx[KPG * NBR];

  const int tid = threadIdx.x;
  const int klin0 = blockIdx.x * KPG;          // 16 consecutive keys, same b
  const int b = klin0 / NKEY;

  // stage small weight-MLP params + neighbor indices
  for (int i = tid; i < DIM * W_HID;  i += 128) s_ww1[i] = ww1[i];
  for (int i = tid; i < W_HID;        i += 128) s_wb1[i] = wb1[i];
  for (int i = tid; i < W_HID*C_MID;  i += 128) s_ww2[i] = ww2[i];
  for (int i = tid; i < C_MID;        i += 128) s_wb2[i] = wb2[i];
  for (int i = tid; i < KPG * NBR;    i += 128) s_idx[i] = idxs[(size_t)klin0 * NBR + i];
  __syncthreads();

  // ---- gather + per-neighbor weight MLP (3 -> 32 -> 16), masked ----
  for (int p = tid; p < KPG * NBR; p += 128) {
    const int q = p >> 5;         // key in group
    const int j = p & 31;         // neighbor
    const int klin = klin0 + q;
    const int idx = s_idx[p];
    const float vf = valid[(size_t)klin * NPT + idx] ? 1.0f : 0.0f;
    const float* pp = points + ((size_t)b * NPT + idx) * 3;
    float rx = pp[0] - keys[klin * 3 + 0];
    float ry = pp[1] - keys[klin * 3 + 1];
    float rz = pp[2] - keys[klin * 3 + 2];
    float hbuf[W_HID];
#pragma unroll
    for (int hh = 0; hh < W_HID; ++hh) {
      float t = fmaf(rx, s_ww1[0 * W_HID + hh],
               fmaf(ry, s_ww1[1 * W_HID + hh],
               fmaf(rz, s_ww1[2 * W_HID + hh], s_wb1[hh])));
      hbuf[hh] = fmaxf(t, 0.0f);
    }
#pragma unroll
    for (int c = 0; c < C_MID; ++c) {
      float m = s_wb2[c];
#pragma unroll
      for (int hh = 0; hh < W_HID; ++hh) m = fmaf(hbuf[hh], s_ww2[hh * C_MID + c], m);
      s_A[q][c][j] = (_Float16)(m * vf);
    }
    const float* fp = feats + ((size_t)b * NPT + idx) * C_IN;
#pragma unroll
    for (int c = 0; c < C_IN; ++c) s_Bt[q][c][j] = (_Float16)(fp[c] * vf);
  }
  __syncthreads();

  const int wave = tid >> 5;
  const int lane = tid & 31;
  const int l15 = lane & 15;
  const int hi  = lane >> 4;

  // ---- per-key einsum: E[q] = m^T(16x32) @ nf(32x16), one WMMA each ----
  for (int q = wave; q < KPG; q += 4) {
    v16h a;
#pragma unroll
    for (int h = 0; h < 16; ++h) {
      int ka = hi * 16 + h;
      a[h] = s_A[q][l15][ka];
    }
    // B fragment: col N=l15, K = hi*16 + h  -> contiguous 16 halves (32B)
    v16h bf = *(const v16h*)&s_Bt[q][l15][hi * 16];
    v8f acc = {};
    acc = __builtin_amdgcn_wmma_f32_16x16x32_f16(false, a, false, bf,
                                                 (short)0, acc, false, false);
#pragma unroll
    for (int r = 0; r < 8; ++r)
      s_E[q][(r + 8 * hi) * C_IN + l15] = (_Float16)acc[r];
  }
  __syncthreads();

  // ---- layer 1: H = relu(E(16x256) @ fw1(256x64) + fb1), wave = N-tile ----
  {
    const int nt = wave;                       // columns nt*16 .. nt*16+15
    v8f acc = {};
#pragma unroll
    for (int kb = 0; kb < 8; ++kb) {           // K blocks of 32
      v16h a;
#pragma unroll
      for (int h = 0; h < 16; ++h) {
        int ka = kb * 32 + (hi * 16 + h);
        a[h] = s_E[l15][ka];
      }
      // pre-swizzled B fragment: one contiguous 32B load per lane
      v16h bf = *(const v16h*)(gw1s + (((size_t)(nt * 8 + kb) * 32 + lane) << 4));
      acc = __builtin_amdgcn_wmma_f32_16x16x32_f16(false, a, false, bf,
                                                   (short)0, acc, false, false);
    }
    const float bias = fb1[nt * 16 + l15];
#pragma unroll
    for (int r = 0; r < 8; ++r)
      s_H[r + 8 * hi][nt * 16 + l15] = (_Float16)fmaxf(acc[r] + bias, 0.0f);
  }
  __syncthreads();

  // ---- layer 2: Out = H(16x64) @ fw2(64x64) + fb2 ----
  {
    const int nt = wave;
    v8f acc = {};
#pragma unroll
    for (int kb = 0; kb < 2; ++kb) {
      v16h a;
#pragma unroll
      for (int h = 0; h < 16; ++h) {
        int ka = kb * 32 + (hi * 16 + h);
        a[h] = s_H[l15][ka];
      }
      v16h bf = *(const v16h*)(gw2s + (((size_t)(nt * 2 + kb) * 32 + lane) << 4));
      acc = __builtin_amdgcn_wmma_f32_16x16x32_f16(false, a, false, bf,
                                                   (short)0, acc, false, false);
    }
    const float bias = fb2[nt * 16 + l15];
#pragma unroll
    for (int r = 0; r < 8; ++r)
      out[(size_t)(klin0 + r + 8 * hi) * C_OUT + nt * 16 + l15] = acc[r] + bias;
  }
}

// =====================================================================
extern "C" void kernel_launch(void* const* d_in, const int* in_sizes, int n_in,
                              void* d_out, int out_size, void* d_ws, size_t ws_size,
                              hipStream_t stream) {
  const float* keys   = (const float*)d_in[0];
  const float* points = (const float*)d_in[1];
  const float* feats  = (const float*)d_in[2];
  const unsigned char* valid = (const unsigned char*)d_in[3];  // jnp bool_
  const float* ww1 = (const float*)d_in[4];
  const float* wb1 = (const float*)d_in[5];
  const float* ww2 = (const float*)d_in[6];
  const float* wb2 = (const float*)d_in[7];
  const float* fw1 = (const float*)d_in[8];
  const float* fb1 = (const float*)d_in[9];
  const float* fw2 = (const float*)d_in[10];
  const float* fb2 = (const float*)d_in[11];

  // workspace layout
  char* ws = (char*)d_ws;
  int* idxs = (int*)ws;                                   // B*K*32 ints = 1 MB
  size_t off = (size_t)BATCH * NKEY * NBR * sizeof(int);
  off = (off + 255) & ~(size_t)255;
  _Float16* gw1s = (_Float16*)(ws + off);                 // 256*64 halfs, swizzled
  off += (size_t)(C_IN * C_MID) * F_HID * sizeof(_Float16);
  off = (off + 255) & ~(size_t)255;
  _Float16* gw2s = (_Float16*)(ws + off);                 // 64*64 halfs, swizzled

  prep_weights<<<64, 256, 0, stream>>>(fw1, fw2, gw1s, gw2s);
  knn_kernel<<<BATCH * NKEY, 256, 0, stream>>>(keys, points, valid, idxs);
  fused_kernel<<<(BATCH * NKEY) / KPG, 128, 0, stream>>>(
      keys, points, feats, valid, ww1, wb1, ww2, wb2,
      gw1s, fb1, gw2s, fb2, idxs, (float*)d_out);
}
